// GAT_custom_13108240187919
// MI455X (gfx1250) — matrix operand order repaired and checked by reading, not compiled, not run on hardware
//
#include <hip/hip_runtime.h>
#include <hip/hip_bf16.h>

// ---------------------------------------------------------------------------
// GAT pipeline for MI455X (gfx1250, wave32, WMMA).
//  - Dense GEMMs (h = relu(x@Wn), xs = x_out@W_src[l]) run as bf16 WMMA
//    v_wmma_f32_16x16x32_bf16 with f32 accumulation (~271 GFLOP total).
//  - GEMM mainloop is double-buffered with CDNA5 async global->LDS copies
//    (global_load_async_to_lds_b128, ASYNCcnt): tile t+1 streams into LDS
//    via the async engine (no VGPR staging) while waves run WMMA on tile t.
//  - xd/el GEMMs are algebraically folded into skinny mat-vecs since the
//    reference only consumes their attention projections (saves ~310 GFLOP).
//  - Attention softmax via atomic segment-max/sum; message scatter via
//    global f32 atomics (agg ~100MB -> mostly L2-resident, 192MB L2).
// ---------------------------------------------------------------------------

#define N_NODES 16384
#define E_EDGES 65536
#define DIN     768
#define HEADS   2
#define CDIM    768
#define HCW     1536            // HEADS*CDIM
#define EHD     128
#define GB      4
#define LLAY    3
#define NCLS    4
#define POOLW   ((LLAY + 1) * HCW)   // 6144
#define LN_EPS  1e-5f

typedef __bf16 bf16_t;
typedef __attribute__((ext_vector_type(16))) __bf16 v16bf;
typedef __attribute__((ext_vector_type(8)))  __bf16 v8bf;
typedef __attribute__((ext_vector_type(8)))  float  v8f;

// ---------------- helpers ----------------

__device__ __forceinline__ unsigned short f32_to_bf16_rne(float f) {
  unsigned int u = __float_as_uint(f);
  unsigned int lsb = (u >> 16) & 1u;
  u += 0x7fffu + lsb;                      // round to nearest even
  return (unsigned short)(u >> 16);
}

__device__ __forceinline__ void atomicMaxFloat(float* addr, float val) {
  // monotonic int encoding of IEEE floats (init must be -inf)
  if (__float_as_int(val) >= 0)
    atomicMax((int*)addr, __float_as_int(val));
  else
    atomicMin((unsigned int*)addr, (unsigned int)__float_as_uint(val));
}

// CDNA5 async global->LDS copy (VGPR-bypassing, tracked by ASYNCcnt).
// Generic shared pointers on amdgcn carry the LDS byte offset in the low
// 32 bits, which is exactly what the VDST operand of the async op wants.
__device__ __forceinline__ void async_copy_b128(unsigned lds_off,
                                                const void* gaddr) {
  asm volatile("global_load_async_to_lds_b128 %0, %1, off"
               :: "v"(lds_off), "v"(gaddr) : "memory");
}

__device__ __forceinline__ void wait_async0() {
#if __has_builtin(__builtin_amdgcn_s_wait_asynccnt)
  __builtin_amdgcn_s_wait_asynccnt(0);
#else
  asm volatile("s_wait_asynccnt 0" ::: "memory");
#endif
}

// ---------------- conversion / fill ----------------

__global__ void cvt_f32_bf16_kernel(const float* __restrict__ src,
                                    unsigned short* __restrict__ dst, long n) {
  long i = (long)blockIdx.x * 256 + threadIdx.x;
  if (i < n) dst[i] = f32_to_bf16_rne(src[i]);
}

__global__ void fill_kernel(float* __restrict__ p, float v, long n) {
  long i = (long)blockIdx.x * 256 + threadIdx.x;
  if (i < n) p[i] = v;
}

__global__ void agg_init_kernel(float* __restrict__ agg,
                                const float* __restrict__ cb) {
  long i = (long)blockIdx.x * 256 + threadIdx.x;
  if (i < (long)N_NODES * HCW) agg[i] = cb[(int)(i % HCW)];
}

// ---------------- WMMA GEMM: C[M,N] = act(A_bf16[M,K] @ B_bf16[K,N] + bias) --
// Block: 256 thr = 8 waves; tile 128x64, BK=32 (exactly one WMMA K per stage).
// Double-buffered LDS, filled by the async engine.
// Requires M%128==0, N%64==0, K%32==0 (true for all call sites here).

__global__ __launch_bounds__(256) void wmma_gemm_kernel(
    const unsigned short* __restrict__ A, const unsigned short* __restrict__ B,
    const float* __restrict__ bias, float* __restrict__ C,
    int M, int Nn, int K, int act) {
  constexpr int BM = 128, BN = 64, BK = 32;
  constexpr int LDA = 40, LDB = 72;        // padded strides, 16B-aligned rows
  __shared__ __align__(16) bf16_t As[2][BM * LDA];
  __shared__ __align__(16) bf16_t Bs[2][BK * LDB];

  const int tid  = threadIdx.x;
  const int lane = tid & 31;
  const int wave = tid >> 5;
  const int wm = wave & 3;                 // 4 waves along M (32 rows each)
  const int wn = wave >> 2;                // 2 waves along N (32 cols each)
  const int m0 = blockIdx.y * BM;
  const int n0 = blockIdx.x * BN;

  // per-thread staging coordinates (3x b128 per thread per K-stage)
  const int ar0 = tid >> 2,         ak0 = (tid & 3) << 3;         // A chunk 0
  const int ar1 = (tid + 256) >> 2, ak1 = ((tid + 256) & 3) << 3; // A chunk 1
  const int brr = tid >> 3,         bnn = (tid & 7) << 3;         // B chunk

  auto stage = [&](int s, int k0) {
    async_copy_b128((unsigned)(size_t)&As[s][ar0 * LDA + ak0],
                    A + (size_t)(m0 + ar0) * K + (k0 + ak0));
    async_copy_b128((unsigned)(size_t)&As[s][ar1 * LDA + ak1],
                    A + (size_t)(m0 + ar1) * K + (k0 + ak1));
    async_copy_b128((unsigned)(size_t)&Bs[s][brr * LDB + bnn],
                    B + (size_t)(k0 + brr) * Nn + (n0 + bnn));
  };

  v8f acc[2][2];
#pragma unroll
  for (int a = 0; a < 2; ++a)
#pragma unroll
    for (int b = 0; b < 2; ++b)
#pragma unroll
      for (int e = 0; e < 8; ++e) acc[a][b][e] = 0.f;

  const int nk = K / BK;
  stage(0, 0);
  wait_async0();
  __syncthreads();

  for (int t = 0; t < nk; ++t) {
    const int cur = t & 1;
    if (t + 1 < nk) stage(cur ^ 1, (t + 1) * BK);   // overlap fill with WMMA

    // A fragment (16x32 bf16): lane L holds row M=L%16;
    //   lanes<16: K = 0..7 (v0-3), 16..23 (v4-7); lanes>=16: +8.
    v16bf afrag[2];
#pragma unroll
    for (int mt = 0; mt < 2; ++mt) {
      const int row = wm * 32 + mt * 16 + (lane & 15);
      const int kb  = (lane >> 4) << 3;
      const bf16_t* p = &As[cur][row * LDA + kb];
      v8bf lo = *(const v8bf*)p;            // K = kb .. kb+7
      v8bf hi = *(const v8bf*)(p + 16);     // K = kb+16 .. kb+23
#pragma unroll
      for (int e = 0; e < 8; ++e) { afrag[mt][e] = lo[e]; afrag[mt][e + 8] = hi[e]; }
    }
    // B fragment (32x16 bf16): lane L holds col N=L%16; K = 0..7, 16..23 (lanes>=16: +8).
    v16bf bfrag[2];
#pragma unroll
    for (int nt = 0; nt < 2; ++nt) {
      const int kb = (lane >> 4) << 3;
      const bf16_t* p = &Bs[cur][kb * LDB + wn * 32 + nt * 16 + (lane & 15)];
#pragma unroll
      for (int e = 0; e < 8; ++e) { bfrag[nt][e] = p[e * LDB]; bfrag[nt][e + 8] = p[(e + 16) * LDB]; }
    }
#pragma unroll
    for (int mt = 0; mt < 2; ++mt)
#pragma unroll
      for (int nt = 0; nt < 2; ++nt)
        acc[mt][nt] = __builtin_amdgcn_wmma_f32_16x16x32_bf16(
            false, afrag[mt], false, bfrag[nt], (short)0, acc[mt][nt],
            false, false);

    if (t + 1 < nk) {
      wait_async0();        // next tile landed in LDS
      __syncthreads();      // all waves done reading cur + see new tile
    }
  }

  // epilogue: C/D layout -> VGPR i: M = i + (lane>=16 ? 8 : 0), N = lane%16
#pragma unroll
  for (int mt = 0; mt < 2; ++mt)
#pragma unroll
    for (int nt = 0; nt < 2; ++nt) {
      const int col   = n0 + wn * 32 + nt * 16 + (lane & 15);
      const int rbase = m0 + wm * 32 + mt * 16 + ((lane >> 4) << 3);
      const float bv  = bias ? bias[col] : 0.f;
#pragma unroll
      for (int i = 0; i < 8; ++i) {
        float v = acc[mt][nt][i] + bv;
        if (act) v = fmaxf(v, 0.f);
        C[(size_t)(rbase + i) * Nn + col] = v;
      }
    }
}

// ---------------- edge preprocess: e = relu(edge_attr @ We + be) ------------

__global__ void edge_pre_kernel(const float* __restrict__ ea,
                                const float* __restrict__ We,
                                const float* __restrict__ be,
                                float* __restrict__ ef) {
  int i = blockIdx.x * 256 + threadIdx.x;
  if (i >= E_EDGES * EHD) return;
  int e = i >> 7, j = i & 127;
  float v = ea[e] * We[j] + be[j];
  ef[i] = v > 0.f ? v : 0.f;
}

// ---------------- attention-vector folding ----------------------------------
// out[i*H+h] = sum_c W[i, h*C+c] * att[h*C+c]   (W is rows x HCW)

__global__ void fold_kernel(const float* __restrict__ W,
                            const float* __restrict__ att,
                            float* __restrict__ out, int rows) {
  int idx = blockIdx.x * 256 + threadIdx.x;
  if (idx >= rows * HEADS) return;
  int i = idx >> 1, h = idx & 1;
  const float* wr = W + (size_t)i * HCW + h * CDIM;
  const float* ar = att + h * CDIM;
  float s = 0.f;
  for (int c = 0; c < CDIM; ++c) s += wr[c] * ar[c];
  out[i * HEADS + h] = s;
}

// s_dst[n,h] = x_out[n,:] @ fd[:,h]
__global__ void node_dot_fold_kernel(const float* __restrict__ X,
                                     const float* __restrict__ f,
                                     float* __restrict__ o) {
  int idx = blockIdx.x * 256 + threadIdx.x;
  if (idx >= N_NODES * HEADS) return;
  int n = idx >> 1, h = idx & 1;
  const float* xr = X + (size_t)n * HCW;
  float s = 0.f;
  for (int k = 0; k < HCW; ++k) s += xr[k] * f[k * HEADS + h];
  o[idx] = s;
}

// s_src[n,h] = sum_c xs[n, h*C+c] * att_src[h*C+c]
__global__ void node_att_dot_kernel(const float* __restrict__ XS,
                                    const float* __restrict__ att,
                                    float* __restrict__ o) {
  int idx = blockIdx.x * 256 + threadIdx.x;
  if (idx >= N_NODES * HEADS) return;
  int n = idx >> 1, h = idx & 1;
  const float* xr = XS + (size_t)n * HCW + h * CDIM;
  const float* ar = att + h * CDIM;
  float s = 0.f;
  for (int c = 0; c < CDIM; ++c) s += xr[c] * ar[c];
  o[idx] = s;
}

// s_edge[e,h] = e_feat[e,:] @ fe[:,h]
__global__ void edge_dot_fold_kernel(const float* __restrict__ EF,
                                     const float* __restrict__ f,
                                     float* __restrict__ o) {
  int idx = blockIdx.x * 256 + threadIdx.x;
  if (idx >= E_EDGES * HEADS) return;
  int e = idx >> 1, h = idx & 1;
  const float* er = EF + (size_t)e * EHD;
  float s = 0.f;
  for (int k = 0; k < EHD; ++k) s += er[k] * f[k * HEADS + h];
  o[idx] = s;
}

// ---------------- segment softmax over incoming edges ------------------------

__global__ void edge_logits_kernel(const float* __restrict__ ssrc,
                                   const float* __restrict__ sdst,
                                   const float* __restrict__ sedge,
                                   const int* __restrict__ ei,
                                   float* __restrict__ a,
                                   float* __restrict__ mmax) {
  int idx = blockIdx.x * 256 + threadIdx.x;
  if (idx >= E_EDGES * HEADS) return;
  int e = idx >> 1, h = idx & 1;
  int r = ei[e], c = ei[E_EDGES + e];
  float v = ssrc[r * HEADS + h] + sdst[c * HEADS + h] + sedge[idx];
  v = v > 0.f ? v : 0.2f * v;               // leaky relu
  a[idx] = v;
  atomicMaxFloat(&mmax[c * HEADS + h], v);
}

__global__ void edge_exp_kernel(const float* __restrict__ a,
                                const float* __restrict__ mmax,
                                const int* __restrict__ ei,
                                float* __restrict__ ex,
                                float* __restrict__ ssum) {
  int idx = blockIdx.x * 256 + threadIdx.x;
  if (idx >= E_EDGES * HEADS) return;
  int e = idx >> 1, h = idx & 1;
  int c = ei[E_EDGES + e];
  float v = expf(a[idx] - mmax[c * HEADS + h]);
  ex[idx] = v;
  atomicAdd(&ssum[c * HEADS + h], v);
}

__global__ void edge_alpha_kernel(const float* __restrict__ ex,
                                  const float* __restrict__ ssum,
                                  const int* __restrict__ ei,
                                  float* __restrict__ alpha) {
  int idx = blockIdx.x * 256 + threadIdx.x;
  if (idx >= E_EDGES * HEADS) return;
  int e = idx >> 1, h = idx & 1;
  int c = ei[E_EDGES + e];
  alpha[idx] = ex[idx] / fmaxf(ssum[c * HEADS + h], 1e-16f);
}

// ---------------- weighted scatter-aggregate --------------------------------

__global__ void scatter_kernel(const float* __restrict__ XS,
                               const float* __restrict__ alpha,
                               const int* __restrict__ ei,
                               float* __restrict__ agg) {
  long idx = (long)blockIdx.x * 256 + threadIdx.x;
  if (idx >= (long)E_EDGES * HCW) return;
  int e = (int)(idx / HCW);
  int j = (int)(idx - (long)e * HCW);
  int h = (j >= CDIM) ? 1 : 0;
  int r = ei[e], c = ei[E_EDGES + e];
  atomicAdd(&agg[(size_t)c * HCW + j], XS[(size_t)r * HCW + j] * alpha[e * HEADS + h]);
}

// ---------------- LayerNorm + PReLU + residual + mean-pool accum ------------

__global__ __launch_bounds__(256) void ln_prelu_pool_kernel(
    const float* __restrict__ agg, const float* __restrict__ g,
    const float* __restrict__ b, const float* __restrict__ pa,
    const int* __restrict__ batch, float* __restrict__ x_out,
    float* __restrict__ xcs, int slot) {
  __shared__ float wsum[8], wsum2[8], bc[2];
  const int n = blockIdx.x;
  const float* row = agg + (size_t)n * HCW;
  float s = 0.f, s2 = 0.f;
  for (int j = threadIdx.x; j < HCW; j += 256) {
    float v = row[j]; s += v; s2 += v * v;
  }
  for (int o = 16; o > 0; o >>= 1) {
    s  += __shfl_down(s, o, 32);
    s2 += __shfl_down(s2, o, 32);
  }
  const int lane = threadIdx.x & 31, wv = threadIdx.x >> 5;
  if (lane == 0) { wsum[wv] = s; wsum2[wv] = s2; }
  __syncthreads();
  if (threadIdx.x == 0) {
    float a = 0.f, a2 = 0.f;
    for (int i = 0; i < 8; ++i) { a += wsum[i]; a2 += wsum2[i]; }
    float mu = a / HCW;
    bc[0] = mu;
    bc[1] = rsqrtf(a2 / HCW - mu * mu + LN_EPS);
  }
  __syncthreads();
  const float mu = bc[0], rs = bc[1];
  const int gi = batch[n];
  for (int j = threadIdx.x; j < HCW; j += 256) {
    float z = (row[j] - mu) * rs * g[j] + b[j];
    z = z > 0.f ? z : pa[j] * z;            // PReLU
    atomicAdd(&xcs[gi * POOLW + slot * HCW + j], z);
    x_out[(size_t)n * HCW + j] += z;        // residual
  }
}

// pool(h) for slot 0
__global__ void pool_accum_kernel(const float* __restrict__ src,
                                  const int* __restrict__ batch,
                                  float* __restrict__ xcs, int slot) {
  long idx = (long)blockIdx.x * 256 + threadIdx.x;
  if (idx >= (long)N_NODES * HCW) return;
  int n = (int)(idx / HCW);
  int j = (int)(idx - (long)n * HCW);
  atomicAdd(&xcs[batch[n] * POOLW + slot * HCW + j], src[idx]);
}

__global__ void count_kernel(const int* __restrict__ batch,
                             float* __restrict__ cnt) {
  int n = blockIdx.x * 256 + threadIdx.x;
  if (n < N_NODES) atomicAdd(&cnt[batch[n]], 1.0f);
}

__global__ void pool_norm_kernel(const float* __restrict__ xcs,
                                 const float* __restrict__ cnt,
                                 float* __restrict__ xc) {
  int idx = blockIdx.x * 256 + threadIdx.x;
  if (idx >= GB * POOLW) return;
  int g = idx / POOLW;
  xc[idx] = xcs[idx] / fmaxf(cnt[g], 1.0f);
}

// ---------------- tiny MLPs (G=4 rows; memory-trivial) ----------------------

__global__ void mlp_kernel(const float* __restrict__ X, const float* __restrict__ W,
                           const float* __restrict__ bvec, float* __restrict__ Y,
                           int Gg, int Kk, int Nn, int act) {
  int idx = blockIdx.x * 256 + threadIdx.x;
  if (idx >= Gg * Nn) return;
  int g = idx / Nn, j = idx - g * Nn;
  float s = bvec[j];
  const float* xr = X + (size_t)g * Kk;
  for (int k = 0; k < Kk; ++k) s += xr[k] * W[(size_t)k * Nn + j];
  if (act) s = fmaxf(s, 0.f);
  Y[idx] = s;
}

__global__ void head_kernel(const float* __restrict__ p2,
                            const float* __restrict__ Wr,
                            const float* __restrict__ br,
                            float* __restrict__ out) {
  __shared__ float hz[GB * NCLS];
  int t = threadIdx.x;
  if (t < GB * NCLS) {
    int g = t / NCLS, c = t - g * NCLS;
    float s = br[c];
    const float* pr = p2 + (size_t)g * HCW;
    for (int k = 0; k < HCW; ++k) s += pr[k] * Wr[k * NCLS + c];
    float h = 1.f / (1.f + expf(-s));
    hz[t] = h;
    out[t] = s;                    // logits
    out[GB * NCLS + t] = h;        // hazards
  }
  __syncthreads();
  if (t < GB) {
    float prod = 1.f;
    for (int c = 0; c < NCLS; ++c) {
      prod *= (1.f - hz[t * NCLS + c]);
      out[2 * GB * NCLS + t * NCLS + c] = prod;  // survival S
    }
  }
}

// ---------------- launch orchestration ---------------------------------------

static inline int cdiv(long a, long b) { return (int)((a + b - 1) / b); }

extern "C" void kernel_launch(void* const* d_in, const int* in_sizes, int n_in,
                              void* d_out, int out_size, void* d_ws, size_t ws_size,
                              hipStream_t stream) {
  (void)in_sizes; (void)n_in; (void)out_size; (void)ws_size;
  const float* x         = (const float*)d_in[0];
  const float* edge_attr = (const float*)d_in[1];
  const int*   ei        = (const int*)  d_in[2];
  const int*   batch     = (const int*)  d_in[3];
  const float* Wn_pre    = (const float*)d_in[4];
  const float* bn_pre    = (const float*)d_in[5];
  const float* We_pre    = (const float*)d_in[6];
  const float* be_pre    = (const float*)d_in[7];
  const float* W_src     = (const float*)d_in[8];
  const float* W_dst     = (const float*)d_in[9];
  const float* att_src   = (const float*)d_in[10];
  const float* att_dst   = (const float*)d_in[11];
  const float* att_edge  = (const float*)d_in[12];
  const float* W_edge    = (const float*)d_in[13];
  const float* conv_bias = (const float*)d_in[14];
  const float* ln_g      = (const float*)d_in[15];
  const float* ln_b      = (const float*)d_in[16];
  const float* prelu_a   = (const float*)d_in[17];
  const float* Wp1       = (const float*)d_in[18];
  const float* bp1       = (const float*)d_in[19];
  const float* Wp2       = (const float*)d_in[20];
  const float* bp2       = (const float*)d_in[21];
  const float* Wr        = (const float*)d_in[22];
  const float* br        = (const float*)d_in[23];
  float* out = (float*)d_out;

  // workspace carve-up (~420 MB)
  char* w = (char*)d_ws; size_t off = 0;
  auto take = [&](size_t bytes) -> void* {
    void* p = w + off; off += (bytes + 255) & ~(size_t)255; return p;
  };
  unsigned short* xb  = (unsigned short*)take((size_t)N_NODES * DIN * 2);
  unsigned short* wb  = (unsigned short*)take((size_t)HCW * HCW * 2);
  unsigned short* xob = (unsigned short*)take((size_t)N_NODES * HCW * 2);
  float* x_out = (float*)take((size_t)N_NODES * HCW * 4);
  float* xs    = (float*)take((size_t)N_NODES * HCW * 4);
  float* agg   = (float*)take((size_t)N_NODES * HCW * 4);
  float* ef    = (float*)take((size_t)E_EDGES * EHD * 4);
  float* sdst  = (float*)take((size_t)N_NODES * HEADS * 4);
  float* ssrc  = (float*)take((size_t)N_NODES * HEADS * 4);
  float* sedge = (float*)take((size_t)E_EDGES * HEADS * 4);
  float* abuf  = (float*)take((size_t)E_EDGES * HEADS * 4);
  float* exbuf = (float*)take((size_t)E_EDGES * HEADS * 4);
  float* alpha = (float*)take((size_t)E_EDGES * HEADS * 4);
  float* mmax  = (float*)take((size_t)N_NODES * HEADS * 4);
  float* ssum  = (float*)take((size_t)N_NODES * HEADS * 4);
  float* fd    = (float*)take((size_t)HCW * HEADS * 4);
  float* fe    = (float*)take((size_t)EHD * HEADS * 4);
  float* xcs   = (float*)take((size_t)GB * POOLW * 4);
  float* xc    = (float*)take((size_t)GB * POOLW * 4);
  float* cnt   = (float*)take((size_t)GB * 4);
  float* p1    = (float*)take((size_t)GB * HCW * 4);
  float* p2    = (float*)take((size_t)GB * HCW * 4);

  const long NHC = (long)N_NODES * HCW;

  // --- preprocess: h = relu(x @ Wn_pre + bn_pre) via bf16 WMMA ---
  cvt_f32_bf16_kernel<<<cdiv((long)N_NODES * DIN, 256), 256, 0, stream>>>(
      x, xb, (long)N_NODES * DIN);
  cvt_f32_bf16_kernel<<<cdiv((long)DIN * HCW, 256), 256, 0, stream>>>(
      Wn_pre, wb, (long)DIN * HCW);
  {
    dim3 grid(HCW / 64, N_NODES / 128);
    wmma_gemm_kernel<<<grid, 256, 0, stream>>>(xb, wb, bn_pre, x_out,
                                               N_NODES, HCW, DIN, 1);
  }
  // --- edge preprocess ---
  edge_pre_kernel<<<cdiv((long)E_EDGES * EHD, 256), 256, 0, stream>>>(
      edge_attr, We_pre, be_pre, ef);

  // --- pooling setup ---
  fill_kernel<<<cdiv(GB * POOLW, 256), 256, 0, stream>>>(xcs, 0.f, GB * POOLW);
  fill_kernel<<<1, 256, 0, stream>>>(cnt, 0.f, GB);
  count_kernel<<<cdiv(N_NODES, 256), 256, 0, stream>>>(batch, cnt);
  pool_accum_kernel<<<cdiv(NHC, 256), 256, 0, stream>>>(x_out, batch, xcs, 0);

  // --- GAT layers ---
  for (int l = 0; l < LLAY; ++l) {
    const float* Wsl  = W_src    + (size_t)l * HCW * HCW;
    const float* Wdl  = W_dst    + (size_t)l * HCW * HCW;
    const float* Wel  = W_edge   + (size_t)l * EHD * HCW;
    const float* asl  = att_src  + (size_t)l * HEADS * CDIM;
    const float* adl  = att_dst  + (size_t)l * HEADS * CDIM;
    const float* ael  = att_edge + (size_t)l * HEADS * CDIM;

    // xs = x_out @ W_src[l]  (bf16 WMMA, 77.3 GFLOP)
    cvt_f32_bf16_kernel<<<cdiv(NHC, 256), 256, 0, stream>>>(x_out, xob, NHC);
    cvt_f32_bf16_kernel<<<cdiv((long)HCW * HCW, 256), 256, 0, stream>>>(
        Wsl, wb, (long)HCW * HCW);
    {
      dim3 grid(HCW / 64, N_NODES / 128);
      wmma_gemm_kernel<<<grid, 256, 0, stream>>>(xob, wb, nullptr, xs,
                                                 N_NODES, HCW, HCW, 0);
    }

    // fold xd/el GEMMs into skinny mat-vecs
    fold_kernel<<<cdiv(HCW * HEADS, 256), 256, 0, stream>>>(Wdl, adl, fd, HCW);
    fold_kernel<<<cdiv(EHD * HEADS, 256), 256, 0, stream>>>(Wel, ael, fe, EHD);
    node_dot_fold_kernel<<<cdiv(N_NODES * HEADS, 256), 256, 0, stream>>>(
        x_out, fd, sdst);
    node_att_dot_kernel<<<cdiv(N_NODES * HEADS, 256), 256, 0, stream>>>(
        xs, asl, ssrc);
    edge_dot_fold_kernel<<<cdiv(E_EDGES * HEADS, 256), 256, 0, stream>>>(
        ef, fe, sedge);

    // segment softmax (kernel boundaries = stream-wide barriers)
    fill_kernel<<<cdiv(N_NODES * HEADS, 256), 256, 0, stream>>>(
        mmax, -__builtin_inff(), N_NODES * HEADS);
    fill_kernel<<<cdiv(N_NODES * HEADS, 256), 256, 0, stream>>>(
        ssum, 0.f, N_NODES * HEADS);
    edge_logits_kernel<<<cdiv(E_EDGES * HEADS, 256), 256, 0, stream>>>(
        ssrc, sdst, sedge, ei, abuf, mmax);
    edge_exp_kernel<<<cdiv(E_EDGES * HEADS, 256), 256, 0, stream>>>(
        abuf, mmax, ei, exbuf, ssum);
    edge_alpha_kernel<<<cdiv(E_EDGES * HEADS, 256), 256, 0, stream>>>(
        exbuf, ssum, ei, alpha);

    // weighted scatter into agg (init with conv_bias)
    agg_init_kernel<<<cdiv(NHC, 256), 256, 0, stream>>>(agg, conv_bias + l * HCW);
    scatter_kernel<<<cdiv((long)E_EDGES * HCW, 256), 256, 0, stream>>>(
        xs, alpha, ei, agg);

    // LayerNorm + PReLU + pool(z) + residual (x_out += z)
    ln_prelu_pool_kernel<<<N_NODES, 256, 0, stream>>>(
        agg, ln_g + l * HCW, ln_b + l * HCW, prelu_a + l * HCW,
        batch, x_out, xcs, l + 1);
  }

  // --- head ---
  pool_norm_kernel<<<cdiv(GB * POOLW, 256), 256, 0, stream>>>(xcs, cnt, xc);
  mlp_kernel<<<cdiv(GB * HCW, 256), 256, 0, stream>>>(xc, Wp1, bp1, p1,
                                                      GB, POOLW, HCW, 1);
  mlp_kernel<<<cdiv(GB * HCW, 256), 256, 0, stream>>>(p1, Wp2, bp2, p2,
                                                      GB, HCW, HCW, 1);
  head_kernel<<<1, 32, 0, stream>>>(p2, Wr, br, out);
}
